// GuidedFusion_79474074845579
// MI455X (gfx1250) — matrix operand order repaired and checked by reading, not compiled, not run on hardware
//
#include <hip/hip_runtime.h>
#include <hip/hip_bf16.h>
#include <stdint.h>

typedef __attribute__((ext_vector_type(8)))  float  v8f;
typedef __attribute__((ext_vector_type(2)))  float  v2f;
typedef __attribute__((ext_vector_type(16))) __bf16 v16bf;

struct __align__(16) u32x4 { uint32_t x, y, z, w; };
struct __align__(16) u32x8 { u32x4 lo, hi; };
struct __align__(16) f32x4 { float x, y, z, w; };
struct __align__(16) bf16x8 { __bf16 v[8]; };

// Load a 32B bf16 WMMA fragment as two 16B chunks (per-lane); callers here use contiguous k = 16*half + s.
__device__ __forceinline__ v16bf load_frag(const __bf16* p0, const __bf16* p1) {
  u32x8 t;
  t.lo = *(const u32x4*)p0;
  t.hi = *(const u32x4*)p1;
  return __builtin_bit_cast(v16bf, t);
}

__device__ __forceinline__ v8f wmma_bf16(v16bf a, v16bf b, v8f c) {
  return __builtin_amdgcn_wmma_f32_16x16x32_bf16(false, a, false, b, (short)0, c,
                                                 false, false);
}

// Async copy 16B/lane from global to LDS (CDNA5 async path, tracked by ASYNCcnt).
__device__ __forceinline__ void async_copy_b128(uint32_t lds_byte_addr,
                                                const void* gaddr) {
  asm volatile("global_load_async_to_lds_b128 %0, %1, off"
               :
               : "v"(lds_byte_addr), "v"(gaddr)
               : "memory");
}

// ---------------------------------------------------------------------------
// Projection: Y[n][d] = sum_c W[d][c] * X[c][n] + bias[d], output bf16 [N][64]
// One wave per 16d x 16n tile, fp32 WMMA 16x16x4 over C=256 (64 k-steps).
// ---------------------------------------------------------------------------
__global__ void __launch_bounds__(128)
proj_kernel(const float* __restrict__ X,   // [B, 256, N]
            const float* __restrict__ W,   // [64, 256]
            const float* __restrict__ bias,// [64]
            __bf16* __restrict__ Y,        // [B, N, 64]
            int N) {
  const int C = 256;
  const int wid  = (blockIdx.x * blockDim.x + threadIdx.x) >> 5;
  const int lane = threadIdx.x & 31;
  const int half = lane >> 4, lm = lane & 15;
  const int ntiles = N >> 4;
  const int b   = wid / (ntiles * 4);
  const int rem = wid % (ntiles * 4);
  const int nt  = rem >> 2;   // spatial tile
  const int dt  = rem & 3;    // d tile (4 x 16 = 64)

  // Running pointers -> strength-reduced addressing in the k-loop.
  const float* wp = W + (size_t)(dt * 16 + lm) * C + 2 * half;
  const float* xp = X + (size_t)b * C * N + (size_t)(2 * half) * N + nt * 16 + lm;

  v8f acc = {0.f, 0.f, 0.f, 0.f, 0.f, 0.f, 0.f, 0.f};
#pragma unroll 4
  for (int c0 = 0; c0 < C; c0 += 4) {
    // A 16x4 f32: VGPR0 K = 2*half, VGPR1 K = 2*half+1 (contiguous pair)
    v2f a = *(const v2f*)wp;
    // B 4x16 f32: lane column n = lm, K = 2*half (+1)
    v2f bb;
    bb.x = xp[0];
    bb.y = xp[N];
    acc = __builtin_amdgcn_wmma_f32_16x16x4_f32(false, a, false, bb, (short)0,
                                                acc, false, false);
    wp += 4;
    xp += (size_t)4 * N;
  }
  // D layout: VGPR r -> row d = r + 8*half, col n = lm
  __bf16* yrow = Y + ((size_t)b * N + nt * 16 + lm) * 64 + dt * 16;
#pragma unroll
  for (int r = 0; r < 8; ++r) {
    int d = r + 8 * half;
    yrow[d] = (__bf16)(acc[r] + bias[dt * 16 + d]);
  }
}

// ---------------------------------------------------------------------------
// Elementwise fp32 -> bf16 (value matrix), 8 elems/thread, b128 in / b128 out
// ---------------------------------------------------------------------------
__global__ void cvt_bf16_kernel(const float* __restrict__ in,
                                __bf16* __restrict__ out, int n) {
  int i = (blockIdx.x * blockDim.x + threadIdx.x) * 8;
  if (i + 8 > n) return;
  f32x4 a = *(const f32x4*)(in + i);
  f32x4 c = *(const f32x4*)(in + i + 4);
  bf16x8 o;
  o.v[0] = (__bf16)a.x; o.v[1] = (__bf16)a.y; o.v[2] = (__bf16)a.z; o.v[3] = (__bf16)a.w;
  o.v[4] = (__bf16)c.x; o.v[5] = (__bf16)c.y; o.v[6] = (__bf16)c.z; o.v[7] = (__bf16)c.w;
  *(bf16x8*)(out + i) = o;
}

// ---------------------------------------------------------------------------
// Fused attention, workgroup-cooperative:
//   - 4 waves / block share one batch; wave w owns q-tile (blk*4+w) and
//     c-tiles {4w..4w+3} of the output.
//   - V h-blocks (16 KB) are staged into LDS once per block with
//     global_load_async_to_lds_b128, double buffered (overlaps compute).
//   - P tiles are shared via LDS (double buffered): each wave computes its own
//     S = Q K^T, exp, rowsum; every wave multiplies all 4 P tiles by its own
//     V c-tiles -> 4x less L2 and LDS read traffic for V.
// ---------------------------------------------------------------------------
#define NL 4096
#define NH 1024
#define CCH 256

#define VBUF_BYTES 16384            // 256c x 32h bf16
#define POFF 32768                  // P: [2][4][16x32 bf16] = 8 KB
#define SOFF 40960                  // scales: [4][16] f32    = 256 B
#define SMEM_BYTES 41216

__global__ void __launch_bounds__(128, 1)
flash_kernel(const __bf16* __restrict__ Q,   // [B, NL, 64]
             const __bf16* __restrict__ K,   // [B, NH, 64]  row h, contiguous d
             const __bf16* __restrict__ V,   // [B, 256, NH] row c, contiguous h
             const float*  __restrict__ low, // [B, 256, NL]
             const float*  __restrict__ gamma,
             float*        __restrict__ out) // [B, 256, NL]
{
  __shared__ __align__(16) char smem[SMEM_BYTES];
  const int w    = threadIdx.x >> 5;        // wave id in block
  const int lane = threadIdx.x & 31;
  const int half = lane >> 4, lm = lane & 15;
  const int qt0 = blockIdx.x * 4;
  const int b   = qt0 >> 8;                 // 256 q-tiles per batch
  const int ql  = ((qt0 + w) & 255) << 4;   // this wave's q rows

  __bf16* Pq  = (__bf16*)(smem + POFF);     // [(buf*4+q)*512] elements
  float*  scl = (float*)(smem + SOFF);      // [q][16]

  const __bf16* Kb = K + (size_t)b * NH * 64;
  const __bf16* Vb = V + (size_t)b * CCH * NH;

  // Global/LDS bases for this thread's async-copy slice of a V h-block:
  // chunk e = i*128 + tid  ->  c = tid/4 + 32*i, seg = tid&3 (16B segments).
  const char* vg_base = (const char*)Vb + (size_t)(threadIdx.x >> 2) * (NH * 2) +
                        (threadIdx.x & 3) * 16;
  const uint32_t vl_base =
      (uint32_t)(uintptr_t)(smem) + (uint32_t)(threadIdx.x >> 2) * 64 +
      (uint32_t)(threadIdx.x & 3) * 16;

  // Kick off stage of V h-block 0 into buffer 0 (overlaps with pass 1).
#pragma unroll
  for (int i = 0; i < 8; ++i)
    async_copy_b128(vl_base + i * 2048, vg_base + (size_t)i * 65536);

  // Q A-fragments (registers, whole tile). A 16x32 bf16 layout:
  // lane row m = lm; K = s*32 + 16*half .. +15 contiguous
  const __bf16* qrow = Q + ((size_t)b * NL + ql + lm) * 64;
  v16bf qa0 = load_frag(qrow +      16 * half, qrow +      16 * half + 8);
  v16bf qa1 = load_frag(qrow + 32 + 16 * half, qrow + 40 + 16 * half);

  // ---- pass 1: row max over all Nh (per-wave, direct K loads) ----
  float rowmax[8];
#pragma unroll
  for (int r = 0; r < 8; ++r) rowmax[r] = -3.0e38f;

  for (int ht = 0; ht < 64; ++ht) {
    const __bf16* krow = Kb + (size_t)(ht * 16 + lm) * 64;
    // B 32x16 bf16 layout: lane col n = lm; K = 16*half .. +15 contiguous
    v16bf kb0 = load_frag(krow +      16 * half, krow +      16 * half + 8);
    v16bf kb1 = load_frag(krow + 32 + 16 * half, krow + 40 + 16 * half);
    v8f s = {0.f, 0.f, 0.f, 0.f, 0.f, 0.f, 0.f, 0.f};
    s = wmma_bf16(qa0, kb0, s);
    s = wmma_bf16(qa1, kb1, s);
#pragma unroll
    for (int r = 0; r < 8; ++r) rowmax[r] = fmaxf(rowmax[r], s[r]);
  }
#pragma unroll
  for (int off = 8; off >= 1; off >>= 1) {
#pragma unroll
    for (int r = 0; r < 8; ++r)
      rowmax[r] = fmaxf(rowmax[r], __shfl_xor(rowmax[r], off, 32));
  }

  // ---- pass 2: streaming P = exp(S - max); O += P @ V (cooperative) ----
  float rowsum[8];
#pragma unroll
  for (int r = 0; r < 8; ++r) rowsum[r] = 0.f;
  v8f o[4][4];                               // [q-tile][owned c-tile]
#pragma unroll
  for (int q = 0; q < 4; ++q)
#pragma unroll
    for (int j = 0; j < 4; ++j)
      o[q][j] = (v8f){0.f, 0.f, 0.f, 0.f, 0.f, 0.f, 0.f, 0.f};

  for (int hb = 0; hb < 32; ++hb) {
    const int cur = hb & 1;
    // --- own S tiles -> exp -> P[cur][w] ---
    __bf16* Pw = Pq + (cur * 4 + w) * 512;
#pragma unroll
    for (int t = 0; t < 2; ++t) {
      const int ht = hb * 2 + t;
      const __bf16* krow = Kb + (size_t)(ht * 16 + lm) * 64;
      v16bf kb0 = load_frag(krow +      16 * half, krow +      16 * half + 8);
      v16bf kb1 = load_frag(krow + 32 + 16 * half, krow + 40 + 16 * half);
      v8f s = {0.f, 0.f, 0.f, 0.f, 0.f, 0.f, 0.f, 0.f};
      s = wmma_bf16(qa0, kb0, s);
      s = wmma_bf16(qa1, kb1, s);
#pragma unroll
      for (int r = 0; r < 8; ++r) {
        float e = __expf(s[r] - rowmax[r]);
        rowsum[r] += e;
        Pw[(r + 8 * half) * 32 + 16 * t + lm] = (__bf16)e;  // C/D -> row major
      }
    }
    // --- V buffer [cur] ready + all P tiles written ---
    asm volatile("s_wait_asynccnt 0x0" ::: "memory");
    __syncthreads();
    // --- prefetch next V h-block into the other buffer ---
    if (hb < 31) {
      const uint32_t lb = vl_base + (1 - cur) * VBUF_BYTES;
      const char*    gb = vg_base + (size_t)(hb + 1) * 64;
#pragma unroll
      for (int i = 0; i < 8; ++i)
        async_copy_b128(lb + i * 2048, gb + (size_t)i * 65536);
    }
    // --- PV: all 4 P tiles x my 4 c-tiles ---
    v16bf pa[4];
#pragma unroll
    for (int q = 0; q < 4; ++q) {
      const __bf16* prow = Pq + (cur * 4 + q) * 512 + lm * 32;
      pa[q] = load_frag(prow + 16 * half, prow + 16 * half + 8);
    }
    const __bf16* Vl = (const __bf16*)(smem + cur * VBUF_BYTES);
#pragma unroll
    for (int j = 0; j < 4; ++j) {
      const int ct = w * 4 + j;
      const __bf16* vrow = Vl + (ct * 16 + lm) * 32 + 16 * half;
      v16bf vb = load_frag(vrow, vrow + 8);
#pragma unroll
      for (int q = 0; q < 4; ++q) o[q][j] = wmma_bf16(pa[q], vb, o[q][j]);
    }
  }

  // ---- share per-row output scales gamma/rowsum through LDS ----
#pragma unroll
  for (int off = 8; off >= 1; off >>= 1) {
#pragma unroll
    for (int r = 0; r < 8; ++r) rowsum[r] += __shfl_xor(rowsum[r], off, 32);
  }
  const float g = gamma[0];
  if (lm == 0) {
#pragma unroll
    for (int r = 0; r < 8; ++r) scl[w * 16 + 8 * half + r] = g / rowsum[r];
  }
  __syncthreads();

  // ---- epilogue: per (q, owned c-tile): scale, LDS transpose, +low, store ----
  float* T = (float*)(smem + POFF + w * 1024);  // per-wave 16x16 f32 tile
  const int cl = lane >> 1;
  const int mc = (lane & 1) << 3;

  for (int q = 0; q < 4; ++q) {
    const size_t obase = (size_t)b * CCH * NL + (size_t)(((qt0 + q) & 255) << 4);
    float sc[8];
#pragma unroll
    for (int r = 0; r < 8; ++r) sc[r] = scl[q * 16 + 8 * half + r];
    for (int j = 0; j < 4; ++j) {
      f32x4 w0, w1;
      w0.x = o[q][j][0] * sc[0]; w0.y = o[q][j][1] * sc[1];
      w0.z = o[q][j][2] * sc[2]; w0.w = o[q][j][3] * sc[3];
      w1.x = o[q][j][4] * sc[4]; w1.y = o[q][j][5] * sc[5];
      w1.z = o[q][j][6] * sc[6]; w1.w = o[q][j][7] * sc[7];
      // writer lane holds c_local = lm, rows m = 8*half + r  -> T[c][m]
      *(f32x4*)(T + lm * 16 + 8 * half)     = w0;
      *(f32x4*)(T + lm * 16 + 8 * half + 4) = w1;
      asm volatile("s_wait_dscnt 0" ::: "memory");
      f32x4 t0 = *(f32x4*)(T + cl * 16 + mc);
      f32x4 t1 = *(f32x4*)(T + cl * 16 + mc + 4);
      const size_t off = obase + (size_t)((w * 4 + j) * 16 + cl) * NL + mc;
      f32x4 l0 = *(const f32x4*)(low + off);
      f32x4 l1 = *(const f32x4*)(low + off + 4);
      f32x4 r0 = {t0.x + l0.x, t0.y + l0.y, t0.z + l0.z, t0.w + l0.w};
      f32x4 r1 = {t1.x + l1.x, t1.y + l1.y, t1.z + l1.z, t1.w + l1.w};
      *(f32x4*)(out + off)     = r0;
      *(f32x4*)(out + off + 4) = r1;
      asm volatile("s_wait_dscnt 0" ::: "memory");
    }
  }
}

// ---------------------------------------------------------------------------
extern "C" void kernel_launch(void* const* d_in, const int* in_sizes, int n_in,
                              void* d_out, int out_size, void* d_ws, size_t ws_size,
                              hipStream_t stream) {
  (void)in_sizes; (void)n_in; (void)out_size; (void)ws_size;
  const float* low   = (const float*)d_in[0];
  const float* high  = (const float*)d_in[1];
  const float* Wq    = (const float*)d_in[2];
  const float* bq    = (const float*)d_in[3];
  const float* Wk    = (const float*)d_in[4];
  const float* bk    = (const float*)d_in[5];
  const float* gamma = (const float*)d_in[6];
  float* out = (float*)d_out;

  const int B = 8, C = 256, Nl = 4096, Nh = 1024, QD = 64;
  __bf16* qbf = (__bf16*)d_ws;                       // B*Nl*64  (4 MB)
  __bf16* kbf = qbf + (size_t)B * Nl * QD;           // B*Nh*64  (1 MB)
  __bf16* vbf = kbf + (size_t)B * Nh * QD;           // B*C*Nh   (4 MB)

  // q projection: 8 * 256 * 4 = 8192 tiles, 4 waves/block
  proj_kernel<<<dim3(B * (Nl / 16) * 4 / 4), dim3(128), 0, stream>>>(low, Wq, bq, qbf, Nl);
  // k projection: 8 * 64 * 4 = 2048 tiles
  proj_kernel<<<dim3(B * (Nh / 16) * 4 / 4), dim3(128), 0, stream>>>(high, Wk, bk, kbf, Nh);
  // value -> bf16
  const int vn = B * C * Nh;
  cvt_bf16_kernel<<<dim3(vn / (256 * 8)), dim3(256), 0, stream>>>(high, vbf, vn);
  // fused attention: 2048 q-tiles, 4 waves (4 q-tiles) per block
  flash_kernel<<<dim3(B * (Nl / 16) / 4), dim3(128), 0, stream>>>(qbf, kbf, vbf, low, gamma, out);
}
